// MambaMoELayer_21036749816513
// MI455X (gfx1250) — matrix operand order repaired and checked by reading, not compiled, not run on hardware
//
#include <hip/hip_runtime.h>
#include <hip/hip_bf16.h>
#include <math.h>

#define BB 4
#define LL 2048
#define DD 1024
#define EE 6
#define FF 1536
#define HH 512
#define NN (BB*LL)   // 8192 tokens

typedef __bf16 bf16_t;
typedef __attribute__((ext_vector_type(16))) __bf16 v16bf;
typedef __attribute__((ext_vector_type(8)))  __bf16 v8bf;
typedef __attribute__((ext_vector_type(8)))  float  v8f;
typedef __attribute__((ext_vector_type(4)))  int    v4i;

// pointer-to-v4i in global (AS1) and LDS (AS3) address spaces
typedef __attribute__((address_space(1))) v4i* g_v4i_p;
typedef __attribute__((address_space(3))) v4i* l_v4i_p;

#if defined(__has_builtin)
#if __has_builtin(__builtin_amdgcn_global_load_async_to_lds_b128)
#define HAVE_ASYNC_LDS 1
#endif
#endif

__device__ __forceinline__ g_v4i_p to_glb(const void* p) {
    return (g_v4i_p)(unsigned long long)p;
}
__device__ __forceinline__ l_v4i_p to_lds(const void* p) {
    return (l_v4i_p)(unsigned int)(unsigned long long)p;
}

__device__ __forceinline__ void wait_async0() {
#ifdef HAVE_ASYNC_LDS
#if __has_builtin(__builtin_amdgcn_s_wait_asynccnt)
    __builtin_amdgcn_s_wait_asynccnt(0);
#else
    asm volatile("s_wait_asynccnt 0" ::: "memory");
#endif
#endif
}

// ---------------------------------------------------------------------------
// Elementwise fp32 -> bf16 convert
// ---------------------------------------------------------------------------
__global__ void cvt_x_kernel(const float* __restrict__ x, bf16_t* __restrict__ xb, int n) {
    int i = blockIdx.x * blockDim.x + threadIdx.x;
    if (i < n) xb[i] = (bf16_t)x[i];
}

// W: [E][R][C] fp32  ->  Wt: [E][C][R] bf16  (transpose + convert)
__global__ void tr_w_kernel(const float* __restrict__ W, bf16_t* __restrict__ Wt, int R, int C) {
    long long i = (long long)blockIdx.x * blockDim.x + threadIdx.x;
    long long tot = (long long)EE * R * C;
    if (i >= tot) return;
    long long rc = (long long)R * C;
    long long e = i / rc;
    long long rem = i - e * rc;
    long long c = rem / R;
    long long r = rem - c * R;
    Wt[i] = (bf16_t)W[e * rc + r * C + c];
}

// ---------------------------------------------------------------------------
// Router: Linear(D->H) -> LayerNorm -> GELU -> Linear(H->E) -> softmax -> top2
// One block handles 8 tokens; one wave finishes each token.
// ---------------------------------------------------------------------------
__global__ __launch_bounds__(256) void router_kernel(
    const float* __restrict__ x, const float* __restrict__ r1_w, const float* __restrict__ r1_b,
    const float* __restrict__ ln_g, const float* __restrict__ ln_b,
    const float* __restrict__ r2_w, const float* __restrict__ r2_b,
    const float* __restrict__ temp, const float* __restrict__ route_bias,
    float* __restrict__ probs)
{
    __shared__ __align__(16) float xs[8 * DD];   // 32 KB
    __shared__ __align__(16) float hs[8 * HH];   // 16 KB
    const int tid = threadIdx.x;
    const int tok0 = blockIdx.x * 8;

    for (int i = tid; i < 8 * DD; i += 256)
        xs[i] = x[(long long)(tok0 + (i >> 10)) * DD + (i & (DD - 1))];
    __syncthreads();

    // h = x @ r1_w + r1_b : thread tid handles columns tid and tid+256
    float a0[8], a1[8];
#pragma unroll
    for (int tk = 0; tk < 8; ++tk) { a0[tk] = 0.f; a1[tk] = 0.f; }
    for (int d = 0; d < DD; ++d) {
        float w0 = r1_w[d * HH + tid];
        float w1 = r1_w[d * HH + tid + 256];
#pragma unroll
        for (int tk = 0; tk < 8; ++tk) {
            float xv = xs[tk * DD + d];
            a0[tk] += xv * w0;
            a1[tk] += xv * w1;
        }
    }
    float bc0 = r1_b[tid], bc1 = r1_b[tid + 256];
#pragma unroll
    for (int tk = 0; tk < 8; ++tk) {
        hs[tk * HH + tid]       = a0[tk] + bc0;
        hs[tk * HH + tid + 256] = a1[tk] + bc1;
    }
    __syncthreads();

    // one wave (32 lanes) per token
    const int tk = tid >> 5, g = tid & 31;
    const float* h = &hs[tk * HH];

    float s = 0.f;
    for (int j = g; j < HH; j += 32) s += h[j];
#pragma unroll
    for (int off = 16; off; off >>= 1) s += __shfl_xor(s, off, 32);
    float mu = s * (1.0f / HH);

    float vs = 0.f;
    for (int j = g; j < HH; j += 32) { float d2 = h[j] - mu; vs += d2 * d2; }
#pragma unroll
    for (int off = 16; off; off >>= 1) vs += __shfl_xor(vs, off, 32);
    float rstd = rsqrtf(vs * (1.0f / HH) + 1e-5f);

    float acc[EE];
#pragma unroll
    for (int e = 0; e < EE; ++e) acc[e] = 0.f;
    for (int j = g; j < HH; j += 32) {
        float hv = (h[j] - mu) * rstd * ln_g[j] + ln_b[j];
        hv = 0.5f * hv * (1.0f + erff(hv * 0.70710678118654752f));   // exact GELU
#pragma unroll
        for (int e = 0; e < EE; ++e) acc[e] += hv * r2_w[j * EE + e];
    }
#pragma unroll
    for (int e = 0; e < EE; ++e)
#pragma unroll
        for (int off = 16; off; off >>= 1) acc[e] += __shfl_xor(acc[e], off, 32);

    if (g == 0) {
        float tinv = 1.0f / temp[0];
        float lg[EE];
#pragma unroll
        for (int e = 0; e < EE; ++e) lg[e] = (acc[e] + r2_b[e]) * tinv + route_bias[e];
        float mx = lg[0];
#pragma unroll
        for (int e = 1; e < EE; ++e) mx = fmaxf(mx, lg[e]);
        float ex[EE], se = 0.f;
#pragma unroll
        for (int e = 0; e < EE; ++e) { ex[e] = expf(lg[e] - mx); se += ex[e]; }
        float inv = 1.0f / se;
        float p[EE];
#pragma unroll
        for (int e = 0; e < EE; ++e) p[e] = ex[e] * inv;
        int i0 = 0;
#pragma unroll
        for (int e = 1; e < EE; ++e) if (p[e] > p[i0]) i0 = e;
        int i1 = (i0 == 0) ? 1 : 0;
#pragma unroll
        for (int e = 0; e < EE; ++e) if (e != i0 && p[e] > p[i1]) i1 = e;
        int n = tok0 + tk;
#pragma unroll
        for (int e = 0; e < EE; ++e)
            probs[(long long)n * EE + e] = (e == i0) ? p[i0] : ((e == i1) ? p[i1] : 0.0f);
    }
}

// ---------------------------------------------------------------------------
// Tiled bf16 WMMA GEMM: C[M x Nc] = A[M x K] * Bt[Nc x K]^T
// Block tile 128x128, 4 waves (2x2), each wave owns 64x64 = 4x4 tiles of 16x16
// via v_wmma_f32_16x16x32_bf16 (K stepped by 32).
// Double-buffered LDS with async global->LDS copies: next K-tile's loads are
// issued right after the barrier and overlap the 16 WMMAs of this K-tile.
// One barrier + one s_wait_asynccnt per iteration.
// MODE 0: epilogue = +b1, GELU/SiLU, write bf16 act.
// MODE 1: epilogue = (+b2) * probs[row][e], accumulate fp32 into out.
// ---------------------------------------------------------------------------
template<int MODE>
__global__ __launch_bounds__(128) void moe_gemm_kernel(
    const bf16_t* __restrict__ A,    // [M x K] row-major
    const bf16_t* __restrict__ Bt,   // [Nc x K] row-major (transposed weight)
    const float*  __restrict__ bias, // [Nc]
    const float*  __restrict__ probs,// [NN x EE]  (MODE 1)
    float*        __restrict__ out,  // [M x Nc] fp32 (MODE 1)
    bf16_t*       __restrict__ actout,//[M x Nc] bf16 (MODE 0)
    int K, int Nc, int expert, int gelu_flag, int first_flag)
{
    __shared__ __align__(16) bf16_t As[2][128 * 40];
    __shared__ __align__(16) bf16_t Bs[2][128 * 40];

    const int tid  = threadIdx.x;          // 0..127
    const int lane = tid & 31;
    const int wave = tid >> 5;             // 0..3
    const int lh   = lane >> 4;            // lane half: 0/1
    const int lm   = lane & 15;
    const int rows0 = blockIdx.y * 128;
    const int cols0 = blockIdx.x * 128;
    const int waveM = (wave & 1) * 64;     // 2 waves along M
    const int waveN = (wave >> 1) * 64;    // 2 waves along N

    v8f acc[4][4];
#pragma unroll
    for (int t = 0; t < 4; ++t)
#pragma unroll
        for (int u = 0; u < 4; ++u) acc[t][u] = v8f{};

    // Each thread owns one 32-bf16 (64B) row segment of each tile per K-step.
    auto issue_tile = [&](int buf, int kb) {
        const bf16_t* ga = A  + (long long)(rows0 + tid) * K + kb * 32;
        const bf16_t* gb = Bt + (long long)(cols0 + tid) * K + kb * 32;
        bf16_t* la = &As[buf][tid * 40];
        bf16_t* lb = &Bs[buf][tid * 40];
#pragma unroll
        for (int i = 0; i < 4; ++i) {
#ifdef HAVE_ASYNC_LDS
            __builtin_amdgcn_global_load_async_to_lds_b128(to_glb(ga + i * 8), to_lds(la + i * 8), 0, 0);
            __builtin_amdgcn_global_load_async_to_lds_b128(to_glb(gb + i * 8), to_lds(lb + i * 8), 0, 0);
#else
            *(v8bf*)(la + i * 8) = *(const v8bf*)(ga + i * 8);
            *(v8bf*)(lb + i * 8) = *(const v8bf*)(gb + i * 8);
#endif
        }
    };

    const int nkb = K >> 5;
    issue_tile(0, 0);

    for (int kb = 0; kb < nkb; ++kb) {
        const int cur = kb & 1;
        // own deposits for `cur` (issued last iteration) retired
        wait_async0();
        // all waves: deposits visible AND everyone done reading buffer cur^1
        __syncthreads();
        if (kb + 1 < nkb) issue_tile(cur ^ 1, kb + 1);   // overlaps WMMAs below

        const bf16_t* as = As[cur];
        const bf16_t* bs = Bs[cur];

        // A fragments: lane half 0 -> K 0..7 / 16..23, half 1 -> K 8..15 / 24..31
        v16bf afr[4], bfr[4];
#pragma unroll
        for (int t = 0; t < 4; ++t) {
            int r = waveM + t * 16 + lm;
            union { v16bf v; v8bf h[2]; } u;
            u.h[0] = *(const v8bf*)&as[r * 40 + lh * 8];
            u.h[1] = *(const v8bf*)&as[r * 40 + lh * 8 + 16];
            afr[t] = u.v;
        }
        // B fragments: lane half 0 -> K 0..7 / 16..23, half 1 -> K 8..15 / 24..31
#pragma unroll
        for (int t = 0; t < 4; ++t) {
            int c = waveN + t * 16 + lm;
            union { v16bf v; v8bf h[2]; } u;
            u.h[0] = *(const v8bf*)&bs[c * 40 + lh * 8];
            u.h[1] = *(const v8bf*)&bs[c * 40 + lh * 8 + 16];
            bfr[t] = u.v;
        }
#pragma unroll
        for (int t = 0; t < 4; ++t)
#pragma unroll
            for (int u = 0; u < 4; ++u)
                acc[t][u] = __builtin_amdgcn_wmma_f32_16x16x32_bf16(
                    false, afr[t], false, bfr[u], (short)0, acc[t][u], false, false);
    }

    // epilogue: C element r of v8f -> row = base + lh*8 + r, col = base + lm
    float bv[4];
#pragma unroll
    for (int u = 0; u < 4; ++u) bv[u] = bias[cols0 + waveN + u * 16 + lm];

#pragma unroll
    for (int t = 0; t < 4; ++t) {
#pragma unroll
        for (int r = 0; r < 8; ++r) {
            int row = rows0 + waveM + t * 16 + lh * 8 + r;
            float p = (MODE == 1) ? probs[(long long)row * EE + expert] : 0.f;
#pragma unroll
            for (int u = 0; u < 4; ++u) {
                int col = cols0 + waveN + u * 16 + lm;
                float v = acc[t][u][r] + bv[u];
                if (MODE == 0) {
                    if (gelu_flag) v = 0.5f * v * (1.0f + erff(v * 0.70710678118654752f));
                    else           v = v / (1.0f + expf(-v));
                    actout[(long long)row * Nc + col] = (bf16_t)v;
                } else {
                    float w = v * p;
                    long long o = (long long)row * Nc + col;
                    if (first_flag) out[o] = w;
                    else            out[o] += w;
                }
            }
        }
    }
}

// ---------------------------------------------------------------------------
extern "C" void kernel_launch(void* const* d_in, const int* in_sizes, int n_in,
                              void* d_out, int out_size, void* d_ws, size_t ws_size,
                              hipStream_t stream) {
    const float* x          = (const float*)d_in[0];
    const float* r1_w       = (const float*)d_in[1];
    const float* r1_b       = (const float*)d_in[2];
    const float* ln_g       = (const float*)d_in[3];
    const float* ln_b       = (const float*)d_in[4];
    const float* r2_w       = (const float*)d_in[5];
    const float* r2_b       = (const float*)d_in[6];
    const float* temp       = (const float*)d_in[7];
    const float* route_bias = (const float*)d_in[8];
    const float* W1         = (const float*)d_in[9];
    const float* b1         = (const float*)d_in[10];
    const float* W2         = (const float*)d_in[11];
    const float* b2         = (const float*)d_in[12];
    float* out = (float*)d_out;

    char* ws = (char*)d_ws;
    size_t off = 0;
    auto walloc = [&](size_t bytes) -> void* {
        void* p = ws + off;
        off += (bytes + 255) & ~(size_t)255;
        return p;
    };
    bf16_t* xb    = (bf16_t*)walloc((size_t)NN * DD * sizeof(bf16_t));
    bf16_t* W1t   = (bf16_t*)walloc((size_t)EE * FF * DD * sizeof(bf16_t)); // [E][F][D]
    bf16_t* W2t   = (bf16_t*)walloc((size_t)EE * DD * FF * sizeof(bf16_t)); // [E][D][F]
    bf16_t* act   = (bf16_t*)walloc((size_t)NN * FF * sizeof(bf16_t));      // reused per expert
    float*  probs = (float*) walloc((size_t)NN * EE * sizeof(float));

    // 1) precision conversion + weight pre-transpose (K-contiguous layouts)
    {
        int n = NN * DD;
        cvt_x_kernel<<<(n + 255) / 256, 256, 0, stream>>>(x, xb, n);
        long long tw = (long long)EE * DD * FF;
        int blocks = (int)((tw + 255) / 256);
        tr_w_kernel<<<blocks, 256, 0, stream>>>(W1, W1t, DD, FF); // -> [E][F][D]
        tr_w_kernel<<<blocks, 256, 0, stream>>>(W2, W2t, FF, DD); // -> [E][D][F]
    }

    // 2) router -> sparse top-2 probs
    router_kernel<<<NN / 8, 256, 0, stream>>>(x, r1_w, r1_b, ln_g, ln_b,
                                              r2_w, r2_b, temp, route_bias, probs);

    // 3) per-expert: GEMM1 (+bias+activation, bf16 act) then GEMM2 (weighted combine)
    for (int e = 0; e < EE; ++e) {
        moe_gemm_kernel<0><<<dim3(FF / 128, NN / 128), 128, 0, stream>>>(
            xb, W1t + (size_t)e * FF * DD, b1 + (size_t)e * FF,
            nullptr, nullptr, act, DD, FF, e, (e % 2 == 0) ? 1 : 0, 0);
        moe_gemm_kernel<1><<<dim3(DD / 128, NN / 128), 128, 0, stream>>>(
            act, W2t + (size_t)e * DD * FF, b2 + (size_t)e * DD,
            probs, out, nullptr, FF, DD, e, 0, (e == 0) ? 1 : 0);
    }
}
